// MultiHeadAttention_54571854463074
// MI455X (gfx1250) — matrix operand order repaired and checked by reading, not compiled, not run on hardware
//
#include <hip/hip_runtime.h>

// ---------------------------------------------------------------------------
// MHA forward for MI455X (gfx1250): bf16 WMMA pipeline.
//   hidden[f32] -> bf16 ; weights[f32] -> bf16 transposed (N-major for B-op)
//   QKV GEMM (wmma bf16, 4x4 register-blocked) -> Q/K [b,h,s,d], V^T [b,h,d,s]
//   flash attention per (b,h), 16-row Q tiles, online softmax
//   proj GEMM (wmma bf16, 4x4 register-blocked) -> f32 out ; proj_bias appended
// ---------------------------------------------------------------------------

typedef __attribute__((ext_vector_type(16))) __bf16 v16bf;
typedef __attribute__((ext_vector_type(8)))  __bf16 v8bf;
typedef __attribute__((ext_vector_type(8)))  float  v8f;

static constexpr int kS    = 2048;
static constexpr int kB    = 2;
static constexpr int kH    = 2048;
static constexpr int kNH   = 16;
static constexpr int kD    = 128;
static constexpr int kH3   = 3 * kH;      // 6144
static constexpr int kRows = kS * kB;     // 4096

__device__ __forceinline__ v8f zero8() {
  v8f r;
#pragma unroll
  for (int i = 0; i < 8; ++i) r[i] = 0.0f;
  return r;
}

__device__ __forceinline__ v16bf comb(v8bf lo, v8bf hi) {
  v16bf r;
#pragma unroll
  for (int i = 0; i < 8; ++i) { r[i] = lo[i]; r[8 + i] = hi[i]; }
  return r;
}

// A operand (16xK tile row per lane%16), interleaved K per ISA 7.12.2:
// elem e -> K = e + 8g (e<8), e + 8 + 8g (e>=8), g = lane>>4
__device__ __forceinline__ v16bf load_a(const __bf16* __restrict__ base, int row,
                                        int ld, int kb, int lane) {
  const __bf16* p = base + (size_t)row * ld + kb + ((lane >> 4) & 1) * 8;
  return comb(*(const v8bf*)p, *(const v8bf*)(p + 16));
}

// B operand (N row per lane%16), interleaved K: elem e -> K = e + 8g (e<8), e + 8 + 8g (e>=8)
__device__ __forceinline__ v16bf load_b(const __bf16* __restrict__ base, int row,
                                        int ld, int kb, int lane) {
  const __bf16* p = base + (size_t)row * ld + kb + ((lane >> 4) & 1) * 8;
  return comb(*(const v8bf*)p, *(const v8bf*)(p + 16));
}

#define WMMA_BF16(Am, Bm, Cm) \
  __builtin_amdgcn_wmma_f32_16x16x32_bf16(false, (Am), false, (Bm), (short)0, (Cm), false, false)

__device__ __forceinline__ float redmax16(float v) {
#pragma unroll
  for (int m = 1; m < 16; m <<= 1) v = fmaxf(v, __shfl_xor(v, m, 32));
  return v;
}
__device__ __forceinline__ float redsum16(float v) {
#pragma unroll
  for (int m = 1; m < 16; m <<= 1) v += __shfl_xor(v, m, 32);
  return v;
}

// ------------------------------ prepass kernels ----------------------------

__global__ void cvt_bf16_kernel(const float* __restrict__ in,
                                __bf16* __restrict__ out, int n) {
  int i = blockIdx.x * blockDim.x + threadIdx.x;
  if (i < n) out[i] = (__bf16)in[i];
}

// in [rows, cols] f32 -> out [cols, rows] bf16
__global__ void transpose_bf16_kernel(const float* __restrict__ in,
                                      __bf16* __restrict__ out,
                                      int rows, int cols) {
  int i = blockIdx.x * blockDim.x + threadIdx.x;
  if (i < rows * cols) {
    int r = i / cols, c = i - r * cols;
    out[(size_t)c * rows + r] = (__bf16)in[i];
  }
}

__global__ void bias_tail_kernel(const float* __restrict__ bias,
                                 float* __restrict__ out, int n) {
  int i = blockIdx.x * blockDim.x + threadIdx.x;
  if (i < n) out[(size_t)kRows * kH + i] = bias[i];
}

// ------------------------------- QKV GEMM ----------------------------------
// C[4096,6144] = A[4096,2048] x W^T-rows + bias; scatter into Q/K/V^T (bf16).
// 4x4 register blocking: each wave owns a 64x64 macro-tile (16 WMMA / k-step).
__global__ __launch_bounds__(256) void qkv_gemm_kernel(
    const __bf16* __restrict__ A, const __bf16* __restrict__ Wt,
    const float* __restrict__ bias,
    __bf16* __restrict__ Qw, __bf16* __restrict__ Kw, __bf16* __restrict__ Vt) {
  const int lane   = threadIdx.x & 31;
  const int wave   = threadIdx.x >> 5;
  const int nmacro = blockIdx.x * 8 + wave;   // 0..95  (x 64 cols)
  const int mmacro = blockIdx.y;              // 0..63  (x 64 rows)
  const int m0 = mmacro * 64 + (lane & 15);
  const int n0 = nmacro * 64 + (lane & 15);

  v8f acc[4][4];
#pragma unroll
  for (int mt = 0; mt < 4; ++mt)
#pragma unroll
    for (int nt = 0; nt < 4; ++nt) acc[mt][nt] = zero8();

  for (int kb = 0; kb < kH; kb += 32) {
    v16bf a[4], b[4];
#pragma unroll
    for (int mt = 0; mt < 4; ++mt) a[mt] = load_a(A, m0 + mt * 16, kH, kb, lane);
#pragma unroll
    for (int nt = 0; nt < 4; ++nt) b[nt] = load_b(Wt, n0 + nt * 16, kH, kb, lane);
#pragma unroll
    for (int mt = 0; mt < 4; ++mt)
#pragma unroll
      for (int nt = 0; nt < 4; ++nt) acc[mt][nt] = WMMA_BF16(a[mt], b[nt], acc[mt][nt]);
  }

#pragma unroll
  for (int nt = 0; nt < 4; ++nt) {
    const int c    = nmacro * 64 + nt * 16 + (lane & 15);
    const int head = c / (3 * kD);
    const int j    = c - head * (3 * kD);
    const float bv = bias[c];
#pragma unroll
    for (int mt = 0; mt < 4; ++mt) {
      const int mBase = mmacro * 64 + mt * 16 + 8 * (lane >> 4);
#pragma unroll
      for (int i = 0; i < 8; ++i) {
        const int r  = mBase + i;
        const int s  = r >> 1;   // row = s*B + b, B=2
        const int bb = r & 1;
        const __bf16 v = (__bf16)(acc[mt][nt][i] + bv);
        const size_t bh = (size_t)(bb * kNH + head);
        if (j < kD) {
          Qw[(bh * kS + s) * kD + j] = v;
        } else if (j < 2 * kD) {
          Kw[(bh * kS + s) * kD + (j - kD)] = v;
        } else {
          Vt[(bh * kD + (j - 2 * kD)) * kS + s] = v;   // V stored transposed
        }
      }
    }
  }
}

// ----------------------------- flash attention -----------------------------
__global__ __launch_bounds__(32) void attn_kernel(
    const __bf16* __restrict__ Qw, const __bf16* __restrict__ Kw,
    const __bf16* __restrict__ Vt, const unsigned char* __restrict__ mask,
    __bf16* __restrict__ ctx) {
  __shared__ __align__(16) __bf16 Pl[16][32];
  const int lane  = threadIdx.x;       // 0..31, one wave per block
  const int mtile = blockIdx.x;        // s/16
  const int bh    = blockIdx.y;        // b*16 + head
  const int bb    = bh >> 4;
  const int head  = bh & 15;
  const __bf16* Qp = Qw + (size_t)bh * kS * kD;
  const __bf16* Kp = Kw + (size_t)bh * kS * kD;
  const __bf16* Vp = Vt + (size_t)bh * kD * kS;
  const unsigned char* mp = mask + (size_t)bb * kS * kS;

  const int mrow = mtile * 16 + (lane & 15);
  v16bf qa[4];
#pragma unroll
  for (int dc = 0; dc < 4; ++dc) qa[dc] = load_a(Qp, mrow, kD, dc * 32, lane);

  v8f O[8];
#pragma unroll
  for (int dt = 0; dt < 8; ++dt) O[dt] = zero8();
  float mr[8], lr[8];
#pragma unroll
  for (int i = 0; i < 8; ++i) { mr[i] = -3.0e38f; lr[i] = 0.0f; }

  const float scale = 0.08838834764831845f;  // 1/sqrt(128)
  const int mBase  = mtile * 16 + 8 * (lane >> 4);
  const int nsteps = (mtile * 16 + 15 + 32) / 32;  // causal column limit

  for (int st = 0; st < nsteps; ++st) {
    const int T0 = st * 32;
    v8f c0 = zero8(), c1 = zero8();
#pragma unroll
    for (int dc = 0; dc < 4; ++dc) {
      v16bf k0 = load_b(Kp, T0 + (lane & 15), kD, dc * 32, lane);
      v16bf k1 = load_b(Kp, T0 + 16 + (lane & 15), kD, dc * 32, lane);
      c0 = WMMA_BF16(qa[dc], k0, c0);
      c1 = WMMA_BF16(qa[dc], k1, c1);
    }
    const int t0 = T0 + (lane & 15);
    const int t1 = t0 + 16;
    float al[8], e0a[8], e1a[8];
#pragma unroll
    for (int i = 0; i < 8; ++i) {
      const int r = mBase + i;
      float s0 = c0[i] * scale;
      float s1 = c1[i] * scale;
      const bool k0m = (t0 > r) || (mp[(size_t)r * kS + t0] != 0);
      const bool k1m = (t1 > r) || (mp[(size_t)r * kS + t1] != 0);
      if (k0m) s0 = -10000.0f;   // TE MASK_FILL (replace, as in reference)
      if (k1m) s1 = -10000.0f;
      const float sm = redmax16(fmaxf(s0, s1));
      const float nm = fmaxf(mr[i], sm);
      const float a  = __expf(mr[i] - nm);
      const float e0 = __expf(s0 - nm);
      const float e1 = __expf(s1 - nm);
      lr[i] = a * lr[i] + redsum16(e0 + e1);
      mr[i] = nm;
      al[i] = a; e0a[i] = e0; e1a[i] = e1;
    }
#pragma unroll
    for (int dt = 0; dt < 8; ++dt)
#pragma unroll
      for (int i = 0; i < 8; ++i) O[dt][i] *= al[i];

    // C/D layout -> A layout for P via LDS (single wave; barriers are cheap)
    __syncthreads();
    {
      const int mloc = 8 * (lane >> 4);
      const int n = lane & 15;
#pragma unroll
      for (int i = 0; i < 8; ++i) {
        Pl[mloc + i][n]      = (__bf16)e0a[i];
        Pl[mloc + i][n + 16] = (__bf16)e1a[i];
      }
    }
    __syncthreads();
    const int mA = lane & 15;
    const int g  = (lane >> 4) & 1;
    const v8bf* rowp = (const v8bf*)(&Pl[mA][0]);
    const v16bf pa = comb(rowp[g], rowp[2 + g]);   // A interleaved-K mapping
#pragma unroll
    for (int dt = 0; dt < 8; ++dt) {
      v16bf vb = load_b(Vp, dt * 16 + (lane & 15), kS, T0, lane);
      O[dt] = WMMA_BF16(pa, vb, O[dt]);
    }
    __syncthreads();
  }

  float inv[8];
#pragma unroll
  for (int i = 0; i < 8; ++i) inv[i] = 1.0f / lr[i];
  const int n = lane & 15;
#pragma unroll
  for (int dt = 0; dt < 8; ++dt) {
#pragma unroll
    for (int i = 0; i < 8; ++i) {
      const int s   = mBase + i;
      const int col = head * kD + dt * 16 + n;
      ctx[((size_t)s * kB + bb) * kH + col] = (__bf16)(O[dt][i] * inv[i]);
    }
  }
}

// ------------------------------ projection GEMM ----------------------------
// Out[4096,2048] = ctx x projW^T-rows, 4x4 register blocking, f32 output.
__global__ __launch_bounds__(256) void proj_gemm_kernel(
    const __bf16* __restrict__ A, const __bf16* __restrict__ Wt,
    float* __restrict__ Out) {
  const int lane   = threadIdx.x & 31;
  const int wave   = threadIdx.x >> 5;
  const int nmacro = blockIdx.x * 8 + wave;   // 0..31
  const int mmacro = blockIdx.y;              // 0..63
  const int m0 = mmacro * 64 + (lane & 15);
  const int n0 = nmacro * 64 + (lane & 15);

  v8f acc[4][4];
#pragma unroll
  for (int mt = 0; mt < 4; ++mt)
#pragma unroll
    for (int nt = 0; nt < 4; ++nt) acc[mt][nt] = zero8();

  for (int kb = 0; kb < kH; kb += 32) {
    v16bf a[4], b[4];
#pragma unroll
    for (int mt = 0; mt < 4; ++mt) a[mt] = load_a(A, m0 + mt * 16, kH, kb, lane);
#pragma unroll
    for (int nt = 0; nt < 4; ++nt) b[nt] = load_b(Wt, n0 + nt * 16, kH, kb, lane);
#pragma unroll
    for (int mt = 0; mt < 4; ++mt)
#pragma unroll
      for (int nt = 0; nt < 4; ++nt) acc[mt][nt] = WMMA_BF16(a[mt], b[nt], acc[mt][nt]);
  }

#pragma unroll
  for (int nt = 0; nt < 4; ++nt) {
    const int c = nmacro * 64 + nt * 16 + (lane & 15);
#pragma unroll
    for (int mt = 0; mt < 4; ++mt) {
      const int mBase = mmacro * 64 + mt * 16 + 8 * (lane >> 4);
#pragma unroll
      for (int i = 0; i < 8; ++i)
        Out[(size_t)(mBase + i) * kH + c] = acc[mt][nt][i];
    }
  }
}

// --------------------------------- launcher --------------------------------
extern "C" void kernel_launch(void* const* d_in, const int* in_sizes, int n_in,
                              void* d_out, int out_size, void* d_ws, size_t ws_size,
                              hipStream_t stream) {
  (void)in_sizes; (void)n_in; (void)out_size; (void)ws_size;
  const float*         hidden = (const float*)d_in[0];
  const unsigned char* amask  = (const unsigned char*)d_in[1];  // bool mask
  const float*         qkv_w  = (const float*)d_in[2];
  const float*         qkv_b  = (const float*)d_in[3];
  const float*         proj_w = (const float*)d_in[4];
  const float*         proj_b = (const float*)d_in[5];
  float* out = (float*)d_out;

  char* ws = (char*)d_ws;                                  // bytes
  __bf16* hb     = (__bf16*)(ws + (size_t)0);              //  16.78 MB
  __bf16* qkvWT  = (__bf16*)(ws + (size_t)16777216);       //  25.17 MB
  __bf16* projWT = (__bf16*)(ws + (size_t)41943040);       //   8.39 MB
  __bf16* Qw     = (__bf16*)(ws + (size_t)50331648);       //  16.78 MB
  __bf16* Kw     = (__bf16*)(ws + (size_t)67108864);       //  16.78 MB
  __bf16* Vt     = (__bf16*)(ws + (size_t)83886080);       //  16.78 MB
  __bf16* ctx    = (__bf16*)(ws + (size_t)100663296);      //  16.78 MB

  {
    int n = kRows * kH;
    cvt_bf16_kernel<<<(n + 255) / 256, 256, 0, stream>>>(hidden, hb, n);
  }
  {
    int n = kH * kH3;
    transpose_bf16_kernel<<<(n + 255) / 256, 256, 0, stream>>>(qkv_w, qkvWT, kH, kH3);
  }
  {
    int n = kH * kH;
    transpose_bf16_kernel<<<(n + 255) / 256, 256, 0, stream>>>(proj_w, projWT, kH, kH);
  }
  // 64-row x 64-col macro-tiles: QKV grid (96/8, 64); proj grid (32/8, 64)
  qkv_gemm_kernel<<<dim3(kH3 / 64 / 8, kRows / 64), 256, 0, stream>>>(
      hb, qkvWT, qkv_b, Qw, Kw, Vt);
  attn_kernel<<<dim3(kS / 16, kB * kNH), 32, 0, stream>>>(Qw, Kw, Vt, amask, ctx);
  proj_gemm_kernel<<<dim3(kH / 64 / 8, kRows / 64), 256, 0, stream>>>(ctx, projWT, out);
  bias_tail_kernel<<<(kH + 255) / 256, 256, 0, stream>>>(proj_b, out, kH);
}
